// Recurrence_21345987461337
// MI455X (gfx1250) — matrix operand order repaired and checked by reading, not compiled, not run on hardware
//
#include <hip/hip_runtime.h>

#define TT 256
#define NNB 1024
#define HHD 256
#define LLR 16
#define CCD 64
#define NAD 16
#define OUTW 278   // a,p,w,v, h[256], probs[16], pp[2]

typedef _Float16 f16;
typedef _Float16 v8h  __attribute__((ext_vector_type(8)));
typedef _Float16 v16h __attribute__((ext_vector_type(16)));
typedef float    v8f  __attribute__((ext_vector_type(8)));

union U16 { v16h v; v8h h[2]; };

// ---------------------------------------------------------------------------
// Generic WMMA GEMM: Y[M x Nn] = A[M x K] @ W[Nn x K]^T (+bias[Nn]) (+relu)
// A, W are f16 row-major; outputs optionally f32 and/or f16.
// One wave per 32x32 output tile: 2x2 grid of v_wmma_f32_16x16x32_f16 with
// independent accumulators (XDL pipelining) and 2x register-level operand
// reuse (2 b128 loads per WMMA instead of 4).
// Requires M % 32 == 0, Nn % 32 == 0, K % 32 == 0.
// ---------------------------------------------------------------------------
__global__ void wmma_gemm_f16(const f16* __restrict__ A, int lda,
                              const f16* __restrict__ W, int K,
                              const float* __restrict__ bias,
                              float* __restrict__ C32, f16* __restrict__ C16,
                              int ldc, int M, int Nn, int relu)
{
  const int wave = threadIdx.x >> 5;
  const int lane = threadIdx.x & 31;
  const int nTilesN = Nn >> 5;
  const long nTiles = (long)(M >> 5) * nTilesN;
  const long tile = (long)blockIdx.x * 8 + wave;
  if (tile >= nTiles) return;                 // wave-uniform exit (EXEC stays all-ones)
  const int tm = (int)(tile / nTilesN);
  const int tn = (int)(tile % nTilesN);
  const int half = lane >> 4;
  const int l15  = lane & 15;

  // A fragment rows: tm*32 + l15 (+16); W rows (= output cols): tn*32 + l15 (+16)
  const f16* Ap0 = A + (size_t)(tm * 32 + l15) * lda;
  const f16* Ap1 = Ap0 + (size_t)16 * lda;
  const f16* Wp0 = W + (size_t)(tn * 32 + l15) * K;
  const f16* Wp1 = Wp0 + (size_t)16 * K;

  v8f acc00 = {}, acc01 = {}, acc10 = {}, acc11 = {};
  for (int kb = 0; kb < K; kb += 32) {
    U16 a0, a1, b0, b1;
    const int kA = kb + half * 8;            // A lane layout: K {base..+7, base+16..+23}
    a0.h[0] = *reinterpret_cast<const v8h*>(Ap0 + kA);
    a0.h[1] = *reinterpret_cast<const v8h*>(Ap0 + kA + 16);
    a1.h[0] = *reinterpret_cast<const v8h*>(Ap1 + kA);
    a1.h[1] = *reinterpret_cast<const v8h*>(Ap1 + kA + 16);
    const int kB = kb + half * 8;            // B lane layout: K {base..+7, base+16..+23}
    b0.h[0] = *reinterpret_cast<const v8h*>(Wp0 + kB);
    b0.h[1] = *reinterpret_cast<const v8h*>(Wp0 + kB + 16);
    b1.h[0] = *reinterpret_cast<const v8h*>(Wp1 + kB);
    b1.h[1] = *reinterpret_cast<const v8h*>(Wp1 + kB + 16);
    // four independent WMMAs -> no accumulator RAW chain inside the step
    acc00 = __builtin_amdgcn_wmma_f32_16x16x32_f16(false, a0.v, false, b0.v,
                                                   (short)0, acc00, false, false);
    acc01 = __builtin_amdgcn_wmma_f32_16x16x32_f16(false, a0.v, false, b1.v,
                                                   (short)0, acc01, false, false);
    acc10 = __builtin_amdgcn_wmma_f32_16x16x32_f16(false, a1.v, false, b0.v,
                                                   (short)0, acc10, false, false);
    acc11 = __builtin_amdgcn_wmma_f32_16x16x32_f16(false, a1.v, false, b1.v,
                                                   (short)0, acc11, false, false);
  }

  for (int cc = 0; cc < 2; ++cc) {
    const int ncol = tn * 32 + cc * 16 + l15;
    const float bn = bias ? bias[ncol] : 0.0f;
    for (int rr = 0; rr < 2; ++rr) {
      const v8f* accp = (rr == 0) ? (cc == 0 ? &acc00 : &acc01)
                                  : (cc == 0 ? &acc10 : &acc11);
      for (int r = 0; r < 8; ++r) {
        const int m = tm * 32 + rr * 16 + r + half * 8;  // C layout: VGPR r -> M=r (+8 hi half)
        float v = (*accp)[r] + bn;
        if (relu) v = fmaxf(v, 0.0f);
        const size_t o = (size_t)m * ldc + ncol;
        if (C32) C32[o] = v;
        if (C16) C16[o] = (f16)v;
      }
    }
  }
}

// ---------------------------------------------------------------------------
// Elementwise / gather / gating kernels
// ---------------------------------------------------------------------------
__global__ void cvt_f32_to_f16(const float* __restrict__ s, f16* __restrict__ d, size_t n)
{
  size_t i = (size_t)blockIdx.x * blockDim.x + threadIdx.x;
  if (i < n) d[i] = (f16)s[i];
}

__global__ void zero_u32(unsigned* __restrict__ p, size_t nwords)
{
  size_t i = (size_t)blockIdx.x * blockDim.x + threadIdx.x;
  if (i < nwords) p[i] = 0u;
}

// Combined post-head weight: rows 0..15 = w_actor, row 16 = w_critic, rest 0.
// bias: rows 0..15 = b_actor, row 16 = b_critic, rest 0.
__global__ void build_post(const float* __restrict__ w_actor, const float* __restrict__ w_critic,
                           const float* __restrict__ b_actor, const float* __restrict__ b_critic,
                           f16* __restrict__ w16, float* __restrict__ b32)
{
  int idx = blockIdx.x * blockDim.x + threadIdx.x;
  if (idx >= 32 * HHD) return;
  int row = idx >> 8, col = idx & 255;
  float v = 0.0f;
  if (row < 16)       v = w_actor[row * HHD + col];
  else if (row == 16) v = w_critic[col];
  w16[idx] = (f16)v;
  if (col == 0) b32[row] = (row < 16) ? b_actor[row] : (row == 16 ? b_critic[0] : 0.0f);
}

// A_f16[(l*N + n), 0:256] = embed[lines[n, l], :]
__global__ void gather_embed(const float* __restrict__ embed, const int* __restrict__ lines,
                             f16* __restrict__ A)
{
  size_t idx = (size_t)blockIdx.x * blockDim.x + threadIdx.x;
  if (idx >= (size_t)LLR * NNB * HHD) return;
  int j = (int)(idx & 255);
  size_t row = idx >> 8;
  int n = (int)(row & (NNB - 1));
  int l = (int)(row >> 10);
  int e = lines[n * LLR + l];
  A[idx] = (f16)embed[(size_t)e * HHD + j];
}

__device__ inline float sigm(float x) { return 1.0f / (1.0f + expf(-x)); }

// Phase-1 GRU gate step. batch b = roll*N + n; input row = ((roll+tt)%L)*N + n.
__global__ void gate1(const float* __restrict__ gi, const float* __restrict__ gh,
                      float* __restrict__ h32, f16* __restrict__ h16, int tt)
{
  size_t idx = (size_t)blockIdx.x * blockDim.x + threadIdx.x;
  if (idx >= (size_t)LLR * NNB * HHD) return;
  int j = (int)(idx & 255);
  size_t b = idx >> 8;
  int n = (int)(b & (NNB - 1));
  int i = (int)(b >> 10);
  size_t grow = ((size_t)((i + tt) & (LLR - 1)) << 10) + n;
  size_t gb = grow * 768;
  size_t hb = b * 768;
  float ir = gi[gb + j],       iz = gi[gb + 256 + j],  in_ = gi[gb + 512 + j];
  float hr = gh[hb + j],       hz = gh[hb + 256 + j],  hn  = gh[hb + 512 + j];
  float r  = sigm(ir + hr);
  float z  = sigm(iz + hz);
  float nn = tanhf(in_ + r * hn);
  size_t ho = b * HHD + j;
  float hnew = (1.0f - z) * nn + z * h32[ho];
  h32[ho] = hnew;
  h16[ho] = (f16)hnew;
}

// Hmem16[(i*N+n), 0:512] = concat(h_f, h_b)
__global__ void build_hmem(const float* __restrict__ hf, const float* __restrict__ hb,
                           f16* __restrict__ hmem)
{
  size_t idx = (size_t)blockIdx.x * blockDim.x + threadIdx.x;
  if (idx >= (size_t)LLR * NNB * 512) return;
  int j = (int)(idx & 511);
  size_t b = idx >> 9;
  float v = (j < 256) ? hf[b * HHD + j] : hb[b * HHD + (j - 256)];
  hmem[idx] = (f16)v;
}

// B[(t*N+n), 0:576] = concat(cond[t,n,0:64], Hmem16[active[t,n]*N + n, 0:512])
__global__ void build_mlp_in(const float* __restrict__ cond, const int* __restrict__ active,
                             const f16* __restrict__ hmem, f16* __restrict__ B)
{
  size_t idx = (size_t)blockIdx.x * blockDim.x + threadIdx.x;
  if (idx >= (size_t)TT * NNB * 576) return;
  int j = (int)(idx % 576);
  size_t row = idx / 576;
  int n = (int)(row & (NNB - 1));
  if (j < CCD) {
    B[idx] = (f16)cond[row * CCD + j];
  } else {
    int a = active[row];
    B[idx] = hmem[((size_t)(a * NNB + n)) * 512 + (j - CCD)];
  }
}

// Phase-2 GRU gate step for one t; also writes h into d_out (stride 278, offset 4).
__global__ void gate2(const f16* __restrict__ gic, const float* __restrict__ gh,
                      const float* __restrict__ hprev, float* __restrict__ hcur,
                      f16* __restrict__ h16row, float* __restrict__ outbase)
{
  size_t idx = (size_t)blockIdx.x * blockDim.x + threadIdx.x;
  if (idx >= (size_t)NNB * HHD) return;
  int j = (int)(idx & 255);
  int n = (int)(idx >> 8);
  size_t gb = (size_t)n * 768;
  float ir = (float)gic[gb + j], iz = (float)gic[gb + 256 + j], in_ = (float)gic[gb + 512 + j];
  float hr = gh[gb + j],         hz = gh[gb + 256 + j],         hn  = gh[gb + 512 + j];
  float r  = sigm(ir + hr);
  float z  = sigm(iz + hz);
  float nn = tanhf(in_ + r * hn);
  float hnew = (1.0f - z) * nn + z * hprev[idx];
  hcur[idx] = hnew;
  h16row[idx] = (f16)hnew;
  outbase[(size_t)n * OUTW + 4 + j] = hnew;
}

// Softmax over actor logits + scatter a/p/w/v/probs/pp into the packed output.
__global__ void finalize(const float* __restrict__ LG, const int* __restrict__ actions,
                         const int* __restrict__ active, const float* __restrict__ p0,
                         const float* __restrict__ pp0, float* __restrict__ out)
{
  size_t row = (size_t)blockIdx.x * blockDim.x + threadIdx.x;
  if (row >= (size_t)TT * NNB) return;
  int n = (int)(row & (NNB - 1));
  const float* lg = LG + row * 32;
  float mx = lg[0];
  for (int i = 1; i < NAD; ++i) mx = fmaxf(mx, lg[i]);
  float e[NAD], s = 0.0f;
  for (int i = 0; i < NAD; ++i) { e[i] = expf(lg[i] - mx); s += e[i]; }
  float inv = 1.0f / s;
  float* o = out + row * OUTW;
  o[0] = (float)actions[row];
  o[1] = p0[n];
  o[2] = (float)active[row];
  o[3] = lg[16];                 // critic (bias already folded in)
  for (int i = 0; i < NAD; ++i) o[260 + i] = e[i] * inv;
  o[276] = pp0[2 * n];
  o[277] = pp0[2 * n + 1];
}

// ---------------------------------------------------------------------------
extern "C" void kernel_launch(void* const* d_in, const int* in_sizes, int n_in,
                              void* d_out, int out_size, void* d_ws, size_t ws_size,
                              hipStream_t stream)
{
  (void)in_sizes; (void)n_in; (void)out_size; (void)ws_size;
  const float* condition = (const float*)d_in[0];
  const int*   active    = (const int*)d_in[1];
  const int*   lines     = (const int*)d_in[2];
  const int*   actions   = (const int*)d_in[3];
  const float* h0        = (const float*)d_in[4];
  const float* p0        = (const float*)d_in[5];
  const float* pp0       = (const float*)d_in[6];
  const float* embed     = (const float*)d_in[7];
  const float* wih_f = (const float*)d_in[8];
  const float* whh_f = (const float*)d_in[9];
  const float* bih_f = (const float*)d_in[10];
  const float* bhh_f = (const float*)d_in[11];
  const float* wih_b = (const float*)d_in[12];
  const float* whh_b = (const float*)d_in[13];
  const float* bih_b = (const float*)d_in[14];
  const float* bhh_b = (const float*)d_in[15];
  const float* f_w0 = (const float*)d_in[16];
  const float* f_b0 = (const float*)d_in[17];
  const float* f_w1 = (const float*)d_in[18];
  const float* f_b1 = (const float*)d_in[19];
  const float* f_w2 = (const float*)d_in[20];
  const float* f_b2 = (const float*)d_in[21];
  const float* c_wih = (const float*)d_in[22];
  const float* c_whh = (const float*)d_in[23];
  const float* c_bih = (const float*)d_in[24];
  const float* c_bhh = (const float*)d_in[25];
  const float* w_critic = (const float*)d_in[26];
  const float* b_critic = (const float*)d_in[27];
  const float* w_actor  = (const float*)d_in[28];
  const float* b_actor  = (const float*)d_in[29];
  float* out = (float*)d_out;

  // ----- workspace bump allocator -----
  char* ws = (char*)d_ws;
  size_t off = 0;
  auto alloc = [&](size_t bytes) -> char* {
    char* p = ws + off;
    off += (bytes + 255) & ~(size_t)255;
    return p;
  };

  const size_t G3 = 3 * HHD * HHD;       // 196608 (3H x H)
  f16* wihf16 = (f16*)alloc(G3 * 2);
  f16* whhf16 = (f16*)alloc(G3 * 2);
  f16* wihb16 = (f16*)alloc(G3 * 2);
  f16* whhb16 = (f16*)alloc(G3 * 2);
  f16* fw016  = (f16*)alloc((size_t)HHD * 576 * 2);
  f16* fw116  = (f16*)alloc((size_t)HHD * HHD * 2);
  f16* fw216  = (f16*)alloc((size_t)HHD * HHD * 2);
  f16* cwih16 = (f16*)alloc(G3 * 2);
  f16* cwhh16 = (f16*)alloc(G3 * 2);
  f16* wpost16 = (f16*)alloc((size_t)32 * HHD * 2);
  float* bpost = (float*)alloc(32 * 4);
  f16* h0_16  = (f16*)alloc((size_t)NNB * HHD * 2);
  f16* Hmem16 = (f16*)alloc((size_t)LLR * NNB * 512 * 2);
  f16* H16ALL = (f16*)alloc((size_t)TT * NNB * HHD * 2);
  f16* X1     = (f16*)alloc((size_t)TT * NNB * HHD * 2);
  f16* X2     = (f16*)alloc((size_t)TT * NNB * HHD * 2);
  float* GH2  = (float*)alloc((size_t)NNB * 768 * 4);
  float* hping = (float*)alloc((size_t)NNB * HHD * 4);
  float* hpong = (float*)alloc((size_t)NNB * HHD * 4);
  float* LG    = (float*)alloc((size_t)TT * NNB * 32 * 4);

  // Union region: phase-1 scratch + MLP input + X0 overlap later with GIC.
  const size_t aBytes  = (size_t)LLR * NNB * HHD * 2;       //  8 MiB
  const size_t giBytes = (size_t)LLR * NNB * 768 * 4;       // 48 MiB
  const size_t h32B    = (size_t)LLR * NNB * HHD * 4;
  const size_t h16B    = (size_t)LLR * NNB * HHD * 2;
  const size_t p1Bytes = aBytes + 2 * giBytes + giBytes + 2 * h32B + 2 * h16B;
  const size_t bBytes  = (size_t)TT * NNB * 576 * 2;        // 288 MiB
  const size_t x0Bytes = (size_t)TT * NNB * HHD * 2;        // 128 MiB
  const size_t gicBytes = (size_t)TT * NNB * 768 * 2;       // 384 MiB
  size_t uBytes = p1Bytes + bBytes + x0Bytes;
  if (gicBytes > uBytes) uBytes = gicBytes;
  char* U = alloc(uBytes);
  f16*   A16  = (f16*)(U);
  float* gi_f = (float*)(U + aBytes);
  float* gi_b = (float*)(U + aBytes + giBytes);
  float* gh1  = (float*)(U + aBytes + 2 * giBytes);
  float* hf32 = (float*)(U + aBytes + 3 * giBytes);
  float* hb32 = (float*)(U + aBytes + 3 * giBytes + h32B);
  f16*   hf16 = (f16*)(U + aBytes + 3 * giBytes + 2 * h32B);
  f16*   hb16 = (f16*)(U + aBytes + 3 * giBytes + 2 * h32B + h16B);
  f16*   Bbuf = (f16*)(U + p1Bytes);
  f16*   X0   = (f16*)(U + p1Bytes + bBytes);
  f16*   GIC  = (f16*)(U);           // overlays dead phase-1/B/X0 data

  // ----- helpers -----
  auto gemm = [&](const f16* A, int lda, const f16* W, int K, const float* bias,
                  float* C32, f16* C16, int ldc, int M, int Nn, int relu) {
    long tiles = (long)(M / 32) * (Nn / 32);
    unsigned blocks = (unsigned)((tiles + 7) / 8);
    wmma_gemm_f16<<<blocks, 256, 0, stream>>>(A, lda, W, K, bias, C32, C16, ldc, M, Nn, relu);
  };
  auto cvt = [&](const float* s, f16* d, size_t n) {
    cvt_f32_to_f16<<<(unsigned)((n + 255) / 256), 256, 0, stream>>>(s, d, n);
  };
  auto zero = [&](void* p, size_t bytes) {
    size_t w = bytes / 4;
    zero_u32<<<(unsigned)((w + 255) / 256), 256, 0, stream>>>((unsigned*)p, w);
  };

  // ----- 0) weight conversions -----
  cvt(wih_f, wihf16, G3);  cvt(whh_f, whhf16, G3);
  cvt(wih_b, wihb16, G3);  cvt(whh_b, whhb16, G3);
  cvt(f_w0, fw016, (size_t)HHD * 576);
  cvt(f_w1, fw116, (size_t)HHD * HHD);
  cvt(f_w2, fw216, (size_t)HHD * HHD);
  cvt(c_wih, cwih16, G3);  cvt(c_whh, cwhh16, G3);
  cvt(h0, h0_16, (size_t)NNB * HHD);
  build_post<<<32, 256, 0, stream>>>(w_actor, w_critic, b_actor, b_critic, wpost16, bpost);

  // ----- 1) task encoder -----
  {
    size_t nE = (size_t)LLR * NNB * HHD;
    gather_embed<<<(unsigned)((nE + 255) / 256), 256, 0, stream>>>(embed, lines, A16);
  }
  const int MB = LLR * NNB;  // 16384
  // hoisted input projections (one GEMM per direction instead of 16)
  gemm(A16, HHD, wihf16, HHD, bih_f, gi_f, nullptr, 768, MB, 768, 0);
  gemm(A16, HHD, wihb16, HHD, bih_b, gi_b, nullptr, 768, MB, 768, 0);
  zero(hf32, (size_t)MB * HHD * 4);  zero(hf16, (size_t)MB * HHD * 2);
  zero(hb32, (size_t)MB * HHD * 4);  zero(hb16, (size_t)MB * HHD * 2);
  const unsigned gBlocks = (unsigned)(((size_t)MB * HHD + 255) / 256);
  for (int t = 0; t < LLR; ++t) {            // forward scan
    gemm(hf16, HHD, whhf16, HHD, bhh_f, gh1, nullptr, 768, MB, 768, 0);
    gate1<<<gBlocks, 256, 0, stream>>>(gi_f, gh1, hf32, hf16, t);
  }
  for (int t = 0; t < LLR; ++t) {            // backward scan (time-reversed input)
    gemm(hb16, HHD, whhb16, HHD, bhh_b, gh1, nullptr, 768, MB, 768, 0);
    gate1<<<gBlocks, 256, 0, stream>>>(gi_b, gh1, hb32, hb16, LLR - 1 - t);
  }
  {
    size_t nH = (size_t)MB * 512;
    build_hmem<<<(unsigned)((nH + 255) / 256), 256, 0, stream>>>(hf32, hb32, Hmem16);
  }

  // ----- 2) batched MLP + GRU input projection over all T*N rows -----
  {
    size_t nB = (size_t)TT * NNB * 576;
    build_mlp_in<<<(unsigned)((nB + 255) / 256), 256, 0, stream>>>(condition, active, Hmem16, Bbuf);
  }
  const int MT = TT * NNB;  // 262144
  gemm(Bbuf, 576, fw016, 576, f_b0, nullptr, X0, HHD, MT, HHD, 1);
  gemm(X0,  HHD, fw116, HHD, f_b1, nullptr, X1, HHD, MT, HHD, 1);
  gemm(X1,  HHD, fw216, HHD, f_b2, nullptr, X2, HHD, MT, HHD, 1);
  gemm(X2,  HHD, cwih16, HHD, c_bih, nullptr, GIC, 768, MT, 768, 0);

  // ----- 3) sequential GRU over T (only h @ c_whh^T remains per step) -----
  const unsigned g2Blocks = (unsigned)(((size_t)NNB * HHD + 255) / 256);
  for (int t = 0; t < TT; ++t) {
    const f16* hprev16 = (t == 0) ? h0_16 : (H16ALL + (size_t)(t - 1) * NNB * HHD);
    gemm(hprev16, HHD, cwhh16, HHD, c_bhh, GH2, nullptr, 768, NNB, 768, 0);
    const float* hprev32 = (t == 0) ? h0 : ((t & 1) ? hping : hpong);
    float* hcur32 = (t & 1) ? hpong : hping;
    gate2<<<g2Blocks, 256, 0, stream>>>(GIC + (size_t)t * NNB * 768, GH2,
                                        hprev32, hcur32,
                                        H16ALL + (size_t)t * NNB * HHD,
                                        out + (size_t)t * NNB * OUTW);
  }

  // ----- 4) actor+critic heads (fused 32-wide GEMM) + softmax/scatter -----
  gemm(H16ALL, HHD, wpost16, HHD, bpost, LG, nullptr, 32, MT, 32, 0);
  finalize<<<(unsigned)(((size_t)MT + 255) / 256), 256, 0, stream>>>(LG, actions, active, p0, pp0, out);
}
